// GPNN_CAD_6906307412304
// MI455X (gfx1250) — matrix operand order repaired and checked by reading, not compiled, not run on hardware
//
#include <hip/hip_runtime.h>
#include <hip/hip_bf16.h>
#include <stdint.h>

// ---------------------------------------------------------------------------
// GPNN_CAD fused implementation for gfx1250 (MI455X).
// Heavy GEMMs (ConvLSTM 1x1 layers, W3 message GEMM) run on
// v_wmma_f32_16x16x32_bf16 with f32 accumulation.  WMMA operands use a
// fragment-major layout (32 contiguous bytes per lane per fragment): A
// fragments are 2x global_load_b128, B fragments 2x ds_load_b128, C results
// packed 8x bf16 -> one ds_store_b128.  All pointwise nonlinearities use the
// native TRANS-pipe ops (v_exp_f32 / v_rcp_f32 / v_log_f32) instead of libm.
// ---------------------------------------------------------------------------

#define BATCH 4
#define NN    256
#define CH    128   // E == F == HID == 128
#define PIX   (NN * NN)            // 65536 pixels per batch
#define TCOLS 64                   // pixels per workgroup tile

#define LOG2E  1.4426950408889634f
#define LOG2E2 2.8853900817779268f
#define LN2    0.6931471805599453f

typedef __attribute__((ext_vector_type(16))) __bf16 v16bf;
typedef __attribute__((ext_vector_type(8)))  float  v8f;
typedef unsigned int uint32;

__device__ __forceinline__ unsigned short f2bf(float f) {
  uint32 b = __builtin_bit_cast(uint32, f);
  b += 0x7FFFu + ((b >> 16) & 1u);            // round-to-nearest-even
  return (unsigned short)(b >> 16);
}
__device__ __forceinline__ float bf2f(uint32 lo16) {
  uint32 b = lo16 << 16;
  return __builtin_bit_cast(float, b);
}
__device__ __forceinline__ uint32 pack2(float a, float b) {
  return (uint32)f2bf(a) | ((uint32)f2bf(b) << 16);
}
// Native TRANS-pipe math: v_exp_f32 (exp2), v_rcp_f32, v_log_f32 (log2)
__device__ __forceinline__ float sigm(float x) {
  return __builtin_amdgcn_rcpf(1.0f + __builtin_amdgcn_exp2f(-LOG2E * x));
}
__device__ __forceinline__ float tanhn(float x) {
  return 1.0f - 2.0f * __builtin_amdgcn_rcpf(1.0f + __builtin_amdgcn_exp2f(LOG2E2 * x));
}
__device__ __forceinline__ float exp_n(float x) {
  return __builtin_amdgcn_exp2f(LOG2E * x);
}
__device__ __forceinline__ float log_n(float x) {
  return LN2 * __builtin_amdgcn_logf(x);
}

// Fragment-major layouts -----------------------------------------------------
// A-fragment (16x32 bf16) of weight matrix [R x 128]: frag = rowtile*4 + ks.
//   Swizzled storage: ((frag*32 + lane)*16 + e) bf16, where
//   lane<16: M=lane, K = ks*32 + e; lane>=16: +16 on K base.
// B-fragment (32x16 bf16) of LDS activation tile [128 K x 64 cols]:
//   frag = ks*4 + ct.  lane<16: N=lane, K=ks*32+e; lane>=16: N=lane-16, K+16.
// Both: one lane's fragment = 32 contiguous bytes.

__device__ __forceinline__ void wmma_bf16(v16bf a, v16bf b, v8f& c) {
  c = __builtin_amdgcn_wmma_f32_16x16x32_bf16(false, a, false, b, (short)0, c,
                                              false, false);
}

// One ConvLSTM 1x1 layer with h0 = c0 = 0 (forget gate dead):
//   x_out = sigmoid(o) * tanh( sigmoid(i) * tanh(g) )
__device__ __forceinline__ void lstm_layer(const v16bf* __restrict__ Wsw,
                                           const float* __restrict__ bias,
                                           const unsigned short* src,
                                           unsigned short* dst) {
  const int lane   = threadIdx.x & 31;
  const int wave   = threadIdx.x >> 5;       // hid tile 0..7
  const int rowAdd = (lane < 16) ? 0 : 8;
  // Preload the wave's 12 A fragments (gates i,g,o rowtiles wave, 16+wave, 24+wave)
  v16bf ai[4], ag[4], ao[4];
#pragma unroll
  for (int ks = 0; ks < 4; ++ks) {
    ai[ks] = Wsw[((wave)      * 4 + ks) * 32 + lane];
    ag[ks] = Wsw[((16 + wave) * 4 + ks) * 32 + lane];
    ao[ks] = Wsw[((24 + wave) * 4 + ks) * 32 + lane];
  }
  const int ksOut   = wave >> 1;
  const int laneOut = (lane & 15) + 16 * (wave & 1);
  for (int ct = 0; ct < 4; ++ct) {
    v8f acc_i = {}; v8f acc_g = {}; v8f acc_o = {};
#pragma unroll
    for (int ks = 0; ks < 4; ++ks) {
      v16bf b = *reinterpret_cast<const v16bf*>(src + ((ks * 4 + ct) * 32 + lane) * 16);
      wmma_bf16(ai[ks], b, acc_i);
      wmma_bf16(ag[ks], b, acc_g);
      wmma_bf16(ao[ks], b, acc_o);
    }
    float xv[8];
#pragma unroll
    for (int r = 0; r < 8; ++r) {
      int c = wave * 16 + r + rowAdd;
      float cc = sigm(acc_i[r] + bias[c]) * tanhn(acc_g[r] + bias[256 + c]);
      xv[r] = sigm(acc_o[r] + bias[384 + c]) * tanhn(cc);
    }
    uint4 q;
    q.x = pack2(xv[0], xv[1]); q.y = pack2(xv[2], xv[3]);
    q.z = pack2(xv[4], xv[5]); q.w = pack2(xv[6], xv[7]);
    *reinterpret_cast<uint4*>(dst + ((ksOut * 4 + ct) * 32 + laneOut) * 16 + rowAdd) = q;
  }
}

// sigmoid(Wo . x2 + bo) head reading the fragment-major LDS tile.
__device__ __forceinline__ void link_head(const unsigned short* x2,
                                          const float* __restrict__ Wo, float bo,
                                          float* __restrict__ Aout, int base) {
  if (threadIdx.x < TCOLS) {
    const int j = threadIdx.x;
    const int ct = j >> 4, Ncol = j & 15;
    float a = bo;
#pragma unroll
    for (int g = 0; g < 8; ++g) {             // channel group c = g*16 + e
      int ks = g >> 1, half = g & 1;
      const uint4* p = reinterpret_cast<const uint4*>(
          x2 + ((ks * 4 + ct) * 32 + Ncol + 16 * half) * 16);
      uint4 q0 = p[0], q1 = p[1];
      uint32 qq[8] = {q0.x, q0.y, q0.z, q0.w, q1.x, q1.y, q1.z, q1.w};
#pragma unroll
      for (int d = 0; d < 8; ++d) {
        int c = g * 16 + 2 * d;
        a += Wo[c]     * bf2f(qq[d] & 0xffffu);
        a += Wo[c + 1] * bf2f(qq[d] >> 16);
      }
    }
    Aout[base + j] = sigm(a);
  }
}

// ------------------- kernel 0: weight convert + swizzle ---------------------
__global__ void k_conv(const float* __restrict__ Wl0, const float* __restrict__ Wl1,
                       const float* __restrict__ Wmsg,
                       unsigned short* Wl0u, unsigned short* Wl1u, unsigned short* W3u) {
  int t = blockIdx.x * 256 + threadIdx.x;     // 65536 threads = 512*128 slots
  int lane = (t >> 4) & 31, e = t & 15, f = t >> 9;
  int rt = f >> 2, ks = f & 3;
  int row = rt * 16 + (lane & 15);
  int K = ks * 32 + ((lane < 16) ? 0 : 16) + e;
  Wl0u[t] = f2bf(Wl0[row * 256 + K]);         // only x-half of [512,256] matters
  Wl1u[t] = f2bf(Wl1[row * 256 + K]);
  if (t < CH * CH) {                          // W3 = W_msg[:,256:384], 32 frags
    W3u[t] = f2bf(Wmsg[row * 384 + 256 + K]);
  }
}

// ------------------------- kernel 1: U = W1*nf, V = W2*nf ------------------
__global__ void k_uv(const float* __restrict__ nf, const float* __restrict__ Wmsg,
                     float* __restrict__ U, float* __restrict__ V) {
  int t = blockIdx.x * 256 + threadIdx.x;     // B*128*256 = 131072
  int b = t >> 15, oi = t & 32767, o = oi >> 8, i = oi & 255;
  const float* x  = nf + b * (CH * NN) + i;   // nf[b,f,i], stride NN in f
  const float* w1 = Wmsg + o * 384;
  const float* w2 = w1 + 128;
  float u = 0.f, v = 0.f;
#pragma unroll 4
  for (int f = 0; f < CH; ++f) { float xf = x[f * NN]; u += w1[f] * xf; v += w2[f] * xf; }
  U[t] = u; V[t] = v;
}

// ------------------------- kernel 2: round-1 link --------------------------
__global__ __launch_bounds__(256) void
k_link1(const float* __restrict__ ef,
        const unsigned short* __restrict__ Wl0u, const float* __restrict__ bl0,
        const unsigned short* __restrict__ Wl1u, const float* __restrict__ bl1,
        const float* __restrict__ Wo, const float* __restrict__ bo,
        float* __restrict__ A1) {
  __shared__ __align__(32) unsigned short buf0[CH * TCOLS];
  __shared__ __align__(32) unsigned short buf1[CH * TCOLS];
  const int b  = blockIdx.x >> 10;
  const int p0 = (blockIdx.x & 1023) * TCOLS;
  const float* efb = ef + (long)b * CH * PIX;
  // stage X tile (fragment-major): thread owns (8-channel group, column)
  for (int idx = threadIdx.x; idx < 1024; idx += 256) {
    int j = idx & 63, k8 = (idx >> 6) * 8;
    float vl[8];
#pragma unroll
    for (int e = 0; e < 8; ++e) vl[e] = efb[(long)(k8 + e) * PIX + p0 + j];
    int ks = k8 >> 5, kkb = k8 & 31;
    int lane2 = (j & 15) + ((kkb >= 16) ? 16 : 0);
    uint4 q;
    q.x = pack2(vl[0], vl[1]); q.y = pack2(vl[2], vl[3]);
    q.z = pack2(vl[4], vl[5]); q.w = pack2(vl[6], vl[7]);
    *reinterpret_cast<uint4*>(buf0 + ((ks * 4 + (j >> 4)) * 32 + lane2) * 16 + (kkb & 15)) = q;
  }
  __syncthreads();
  lstm_layer((const v16bf*)Wl0u, bl0, buf0, buf1);
  __syncthreads();
  lstm_layer((const v16bf*)Wl1u, bl1, buf1, buf0);
  __syncthreads();
  link_head(buf0, Wo, bo[0], A1, b * PIX + p0);
}

// ------------------------- kernel 3: round-2 link --------------------------
// X2[b,c,v,w] = (U[b,c,w] + V[b,c,v] + (W3*ef)[b,c,w,v] + b_msg[c]) * A1[b,w,v]
__global__ __launch_bounds__(256) void
k_link2(const float* __restrict__ ef, const float* __restrict__ U,
        const float* __restrict__ V, const float* __restrict__ bmsg,
        const float* __restrict__ A1, const unsigned short* __restrict__ W3u,
        const unsigned short* __restrict__ Wl0u, const float* __restrict__ bl0,
        const unsigned short* __restrict__ Wl1u, const float* __restrict__ bl1,
        const float* __restrict__ Wo, const float* __restrict__ bo,
        float* __restrict__ A2) {
  __shared__ __align__(32) unsigned short buf0[CH * TCOLS];
  __shared__ __align__(32) unsigned short buf1[CH * TCOLS];
  const int b  = blockIdx.x >> 10;
  const int p0 = (blockIdx.x & 1023) * TCOLS;
  const int v  = p0 >> 8;            // tile is one v-row, 64 consecutive w
  const int w0 = p0 & 255;
  const float* efb = ef + (long)b * CH * PIX;
  // stage Bt[k][j] = ef[b,k,w0+j,v] (transposed spatial access), fragment-major
  for (int idx = threadIdx.x; idx < 1024; idx += 256) {
    int j = idx & 63, k8 = (idx >> 6) * 8;
    float vl[8];
#pragma unroll
    for (int e = 0; e < 8; ++e) vl[e] = efb[(long)(k8 + e) * PIX + (w0 + j) * NN + v];
    int ks = k8 >> 5, kkb = k8 & 31;
    int lane2 = (j & 15) + ((kkb >= 16) ? 16 : 0);
    uint4 q;
    q.x = pack2(vl[0], vl[1]); q.y = pack2(vl[2], vl[3]);
    q.z = pack2(vl[4], vl[5]); q.w = pack2(vl[6], vl[7]);
    *reinterpret_cast<uint4*>(buf1 + ((ks * 4 + (j >> 4)) * 32 + lane2) * 16 + (kkb & 15)) = q;
  }
  __syncthreads();
  const int lane = threadIdx.x & 31;
  const int wave = threadIdx.x >> 5;
  const int rowAdd = (lane < 16) ? 0 : 8;
  const v16bf* W3sw = (const v16bf*)W3u;
  v16bf w3f[4];
#pragma unroll
  for (int ks = 0; ks < 4; ++ks) w3f[ks] = W3sw[(wave * 4 + ks) * 32 + lane];
  const float* Ub  = U  + b * (CH * NN);
  const float* Vb  = V  + b * (CH * NN);
  const float* A1b = A1 + b * PIX;
  const int ksOut   = wave >> 1;
  const int laneOut = (lane & 15) + 16 * (wave & 1);
  for (int ct = 0; ct < 4; ++ct) {
    v8f acc = {};
#pragma unroll
    for (int ks = 0; ks < 4; ++ks) {
      v16bf bfr = *reinterpret_cast<const v16bf*>(buf1 + ((ks * 4 + ct) * 32 + lane) * 16);
      wmma_bf16(w3f[ks], bfr, acc);
    }
    const int j = ct * 16 + (lane & 15);
    const int w = w0 + j;
    const float a1 = A1b[w * NN + v];
    float xv[8];
#pragma unroll
    for (int r = 0; r < 8; ++r) {
      int c = wave * 16 + r + rowAdd;
      xv[r] = (acc[r] + Ub[c * NN + w] + Vb[c * NN + v] + bmsg[c]) * a1;
    }
    uint4 q;
    q.x = pack2(xv[0], xv[1]); q.y = pack2(xv[2], xv[3]);
    q.z = pack2(xv[4], xv[5]); q.w = pack2(xv[6], xv[7]);
    *reinterpret_cast<uint4*>(buf0 + ((ksOut * 4 + ct) * 32 + laneOut) * 16 + rowAdd) = q;
  }
  __syncthreads();
  lstm_layer((const v16bf*)Wl0u, bl0, buf0, buf1);
  __syncthreads();
  lstm_layer((const v16bf*)Wl1u, bl1, buf1, buf0);
  __syncthreads();
  link_head(buf0, Wo, bo[0], A2, b * PIX + p0);
}

// ------------------------- kernel 4a: S_A[b,i] = sum_w A2[b,i,w] -----------
__global__ void k_sa(const float* __restrict__ A2, float* __restrict__ SA) {
  int row  = blockIdx.x * 8 + (threadIdx.x >> 5);   // B*N = 1024 rows
  int lane = threadIdx.x & 31;
  const float* a = A2 + (long)row * NN;
  float s = 0.f;
  for (int w = lane; w < NN; w += 32) s += a[w];
  for (int off = 16; off; off >>= 1) s += __shfl_xor(s, off, 32);
  if (lane == 0) SA[row] = s;
}

// ------------------------- kernel 4b: Q[b,c,i] = sum_w ef[b,c,i,w]*A2[b,i,w]
__global__ void k_q(const float* __restrict__ ef, const float* __restrict__ A2,
                    float* __restrict__ Q) {
  int row  = blockIdx.x * 8 + (threadIdx.x >> 5);   // B*128*256 rows
  int lane = threadIdx.x & 31;
  int b = row >> 15, ci = row & 32767, c = ci >> 8, i = ci & 255;
  const float* e = ef + (long)b * CH * PIX + (long)c * PIX + i * NN;
  const float* a = A2 + (long)b * PIX + i * NN;
  float s = 0.f;
  for (int w = lane; w < NN; w += 32) s += e[w] * a[w];
  for (int off = 16; off; off >>= 1) s += __shfl_xor(s, off, 32);
  if (lane == 0) Q[row] = s;
}

// ------------------------- kernel 5: m_sum recombine ------------------------
__global__ void k_msum(const float* __restrict__ U, const float* __restrict__ V,
                       const float* __restrict__ A2, const float* __restrict__ Q,
                       const float* __restrict__ SA, const float* __restrict__ Wmsg,
                       const float* __restrict__ bmsg, float* __restrict__ msum) {
  int t = blockIdx.x * 256 + threadIdx.x;           // B*128*256
  int b = t >> 15, oi = t & 32767, o = oi >> 8, i = oi & 255;
  const float* Vb = V + b * (CH * NN) + o * NN;
  const float* a  = A2 + (long)b * PIX + i * NN;
  float s = (U[t] + bmsg[o]) * SA[b * NN + i];
#pragma unroll 4
  for (int w = 0; w < NN; ++w) s += Vb[w] * a[w];
  const float* Qb = Q + b * (CH * NN);
  const float* W3 = Wmsg + o * 384 + 256;
#pragma unroll 4
  for (int c = 0; c < CH; ++c) s += W3[c] * Qb[c * NN + i];
  msum[t] = s;
}

// ------------------------- kernel 6: GRU + log_softmax readout -------------
__global__ __launch_bounds__(128) void
k_gru(const float* __restrict__ msum, const float* __restrict__ nf,
      const float* __restrict__ Wih0, const float* __restrict__ Whh0,
      const float* __restrict__ bih0, const float* __restrict__ bhh0,
      const float* __restrict__ Wih1, const float* __restrict__ Whh1,
      const float* __restrict__ bih1, const float* __restrict__ bhh1,
      const float* __restrict__ Wr0, const float* __restrict__ br0,
      const float* __restrict__ Wr1, const float* __restrict__ br1,
      float* __restrict__ pred) {
  __shared__ float xs[CH], hs[CH], hn[CH], lg[12], lse[1];
  const int bn = blockIdx.x;                  // B*N = 1024 blocks
  const int b = bn >> 8, n = bn & 255;
  const int o = threadIdx.x;
  xs[o] = msum[b * (CH * NN) + o * NN + n];
  hs[o] = nf[b * (CH * NN) + o * NN + n];
  __syncthreads();
  const float* Wih = n ? Wih1 : Wih0;  const float* Whh = n ? Whh1 : Whh0;
  const float* bih = n ? bih1 : bih0;  const float* bhh = n ? bhh1 : bhh0;
  float gir = bih[o], giz = bih[128 + o], gin = bih[256 + o];
  float ghr = bhh[o], ghz = bhh[128 + o], ghn = bhh[256 + o];
#pragma unroll 4
  for (int f = 0; f < CH; ++f) {
    float xf = xs[f], hf = hs[f];
    gir += Wih[o * CH + f] * xf;
    giz += Wih[(128 + o) * CH + f] * xf;
    gin += Wih[(256 + o) * CH + f] * xf;
    ghr += Whh[o * CH + f] * hf;
    ghz += Whh[(128 + o) * CH + f] * hf;
    ghn += Whh[(256 + o) * CH + f] * hf;
  }
  float r   = sigm(gir + ghr);
  float z   = sigm(giz + ghz);
  float nnv = tanhn(gin + r * ghn);
  hn[o] = (1.f - z) * nnv + z * hs[o];
  __syncthreads();
  const int K = n ? 12 : 10;
  const float* Wr = n ? Wr1 : Wr0;  const float* br = n ? br1 : br0;
  if (o < K) {
    float s = br[o];
#pragma unroll 4
    for (int f = 0; f < CH; ++f) s += Wr[o * CH + f] * hn[f];
    lg[o] = s;
  }
  __syncthreads();
  if (o == 0) {
    float mx = lg[0];
    for (int k = 1; k < K; ++k) mx = fmaxf(mx, lg[k]);
    float se = 0.f;
    for (int k = 0; k < K; ++k) se += exp_n(lg[k] - mx);
    lse[0] = mx + log_n(se);
  }
  __syncthreads();
  if (o < 12) {
    float val = (o < K) ? (lg[o] - lse[0]) : 0.f;
    pred[(b * NN + n) * 12 + o] = val;
  }
}

// ---------------------------------------------------------------------------
extern "C" void kernel_launch(void* const* d_in, const int* in_sizes, int n_in,
                              void* d_out, int out_size, void* d_ws, size_t ws_size,
                              hipStream_t stream) {
  (void)in_sizes; (void)n_in; (void)out_size; (void)ws_size;
  const float* ef    = (const float*)d_in[0];
  const float* nf    = (const float*)d_in[1];
  // d_in[2] adj_mat, d_in[3] node_labels, d_in[4] args are unused
  const float* Wl0   = (const float*)d_in[5];
  const float* bl0   = (const float*)d_in[6];
  const float* Wl1   = (const float*)d_in[7];
  const float* bl1   = (const float*)d_in[8];
  const float* Wo    = (const float*)d_in[9];
  const float* bo    = (const float*)d_in[10];
  const float* Wmsg  = (const float*)d_in[11];
  const float* bmsg  = (const float*)d_in[12];
  const float* Wih0  = (const float*)d_in[13];
  const float* Whh0  = (const float*)d_in[14];
  const float* bih0  = (const float*)d_in[15];
  const float* bhh0  = (const float*)d_in[16];
  const float* Wih1  = (const float*)d_in[17];
  const float* Whh1  = (const float*)d_in[18];
  const float* bih1  = (const float*)d_in[19];
  const float* bhh1  = (const float*)d_in[20];
  const float* Wr0   = (const float*)d_in[21];
  const float* br0   = (const float*)d_in[22];
  const float* Wr1   = (const float*)d_in[23];
  const float* br1   = (const float*)d_in[24];

  float* outA  = (float*)d_out;                 // A2 -> first B*N*N floats
  float* pred  = outA + BATCH * PIX;            // pred -> B*N*12 floats

  char* ws = (char*)d_ws;
  float* A1   = (float*)(ws);                                  // 1 MB
  float* U    = (float*)(ws + 1048576);                        // 512 KB
  float* V    = (float*)(ws + 1572864);                        // 512 KB
  float* Q    = (float*)(ws + 2097152);                        // 512 KB
  float* SA   = (float*)(ws + 2621440);                        // 4 KB
  float* msum = (float*)(ws + 2625536);                        // 512 KB
  unsigned short* Wl0u = (unsigned short*)(ws + 3149824);      // 128 KB (32B-aligned)
  unsigned short* Wl1u = (unsigned short*)(ws + 3280896);      // 128 KB
  unsigned short* W3u  = (unsigned short*)(ws + 3411968);      // 32 KB

  k_conv<<<256, 256, 0, stream>>>(Wl0, Wl1, Wmsg, Wl0u, Wl1u, W3u);
  k_uv<<<512, 256, 0, stream>>>(nf, Wmsg, U, V);
  k_link1<<<BATCH * PIX / TCOLS, 256, 0, stream>>>(ef, Wl0u, bl0, Wl1u, bl1, Wo, bo, A1);
  k_link2<<<BATCH * PIX / TCOLS, 256, 0, stream>>>(ef, U, V, bmsg, A1, W3u,
                                                   Wl0u, bl0, Wl1u, bl1, Wo, bo, outA);
  k_sa<<<128, 256, 0, stream>>>(outA, SA);
  k_q<<<16384, 256, 0, stream>>>(ef, outA, Q);
  k_msum<<<512, 256, 0, stream>>>(U, V, outA, Q, SA, Wmsg, bmsg, msum);
  k_gru<<<BATCH * NN, 128, 0, stream>>>(msum, nf, Wih0, Whh0, bih0, bhh0,
                                        Wih1, Whh1, bih1, bhh1, Wr0, br0, Wr1, br1, pred);
}
